// DSTP_RNN_11690900980422
// MI455X (gfx1250) — compile-verified, matrix operand order checked by reading, not run on hardware
//
#include <hip/hip_runtime.h>
#include <hip/hip_bf16.h>

typedef __attribute__((ext_vector_type(16))) __bf16 v16bf;
typedef __attribute__((ext_vector_type(8)))  float  v8f;

#define TB   16     // batch tile (WMMA M)
#define HH   128    // hidden
#define G4   512    // 4*H gates
#define TE   64     // encoder length
#define TDTOT 24    // decoder steps (T_DEC + 6)

// ---- LDS carve (bytes). Total 315,712 B < 320KB/WGP ----
#define OFF_WHH   0                     // bf16[512][128] = 131072
#define OFF_WIH   131072                // bf16[512][128] = 131072 (phase A uses [512][32])
#define OFF_WIX   (OFF_WIH + 32768)     // f32[16][17][64] = 69632 (inside WIH slot, phase A only)
#define OFF_BSUM  262144                // f32[512]
#define OFF_WLAST 264192                // f32[512]
#define OFF_H     266240                // f32[16][128]
#define OFF_C     274432                // f32[16][128]
#define OFF_HBF   282624                // bf16[16][128]
#define OFF_XABF  286720                // bf16[16][128]
#define OFF_E     290816                // f32[16][128]
#define OFF_ATT   299008                // f32[16][132]
#define OFF_LS    307456                // f32[16]
#define OFF_CBF   307520                // bf16[16][128] = 4096
#define OFF_E2    311616                // f32[16][64]   = 4096
#define SMEM_SZ   315712

struct P {
  const float *inp, *label, *Wih1, *Whh1, *bih1, *bhh1, *Wih2, *Whh2, *bih2, *bhh2,
              *Wihd, *Whhd, *bihd, *bhhd, *Wi_w, *Wi_b, *We_w, *Vd_w, *Vd_b,
              *Wi2_w, *Wi2_b, *We2_w, *Vd2_w, *Vd2_b, *Wx_w, *Wx_b, *Wh_w, *V_w, *V_b,
              *reg_w, *reg_b;
  float *mid, *wi2x, *fin, *wxf, *out;
};

__device__ __forceinline__ float sigm(float x) { return 1.0f / (1.0f + __expf(-x)); }

__device__ __forceinline__ float htanh(float x) {
#if __has_builtin(__builtin_amdgcn_tanhf)
  return __builtin_amdgcn_tanhf(x);       // v_tanh_f32 (CDNA5 hardware transcendental)
#else
  return tanhf(x);
#endif
}

// A/B fragment loader: 16-bit, 16x32 (MxK) per CDNA5 layout:
// lanes 0-15: elems 0..7 -> K=k0+0..7, elems 8..15 -> K=k0+16..23
// lanes 16-31: +8 on both halves. row = row0 + (lane&15).
__device__ __forceinline__ v16bf ld_frag_bf(const __bf16* base, int stride, int row0, int k0, int lane) {
  const __bf16* pp = base + (row0 + (lane & 15)) * stride + k0 + ((lane >> 4) << 3);
  v16bf v;
#pragma unroll
  for (int j = 0; j < 8; ++j) v[j] = pp[j];
#pragma unroll
  for (int j = 0; j < 8; ++j) v[j + 8] = pp[j + 16];
  return v;
}

__device__ __forceinline__ v16bf ld_frag_f32(const float* base, int stride, int row0, int k0, int lane) {
  const float* pp = base + (size_t)(row0 + (lane & 15)) * stride + k0 + ((lane >> 4) << 3);
  v16bf v;
#pragma unroll
  for (int j = 0; j < 8; ++j) v[j] = (__bf16)pp[j];
#pragma unroll
  for (int j = 0; j < 8; ++j) v[j + 8] = (__bf16)pp[j + 16];
  return v;
}

__device__ __forceinline__ v8f wmma32(v16bf a, v16bf b, v8f c) {
  return __builtin_amdgcn_wmma_f32_16x16x32_bf16(false, a, false, b, (short)0, c, false, false);
}

// e = [h,c] @ W.T for the encoder phases (N = 64): waves 0-3 take the h-half of K,
// waves 4-7 the c-half; partials land in e_s / e2_s and are summed by the consumer.
__device__ __forceinline__ void attn_e_enc(const float* W, const __bf16* hbf, const __bf16* cbf,
                                           float* e_s, float* e2_s, int lane, int wave)
{
  const int nl = lane & 15, mb = (lane >> 4) << 3;
  const int ntile = wave & 3, khalf = wave >> 2;
  const __bf16* src = khalf ? cbf : hbf;
  v8f a;
#pragma unroll
  for (int j = 0; j < 8; ++j) a[j] = 0.f;
#pragma unroll
  for (int kt = 0; kt < 4; ++kt)
    a = wmma32(ld_frag_bf(src, HH, 0, kt * 32, lane),
               ld_frag_f32(W + khalf * HH, 256, ntile * 16, kt * 32, lane), a);
  float* dst = khalf ? e2_s : e_s;
  const int stride = khalf ? TE : HH;
#pragma unroll
  for (int j = 0; j < 8; ++j) dst[(mb + j) * stride + ntile * 16 + nl] = a[j];
}

// e = [h,c] @ Wh.T for the decoder (N = 128): one N-tile per wave, full K = 256.
__device__ __forceinline__ void attn_e_dec(const float* W, const __bf16* hbf, const __bf16* cbf,
                                           float* e_s, int lane, int wave)
{
  const int nl = lane & 15, mb = (lane >> 4) << 3;
  v8f a;
#pragma unroll
  for (int j = 0; j < 8; ++j) a[j] = 0.f;
#pragma unroll
  for (int kt = 0; kt < 4; ++kt)
    a = wmma32(ld_frag_bf(hbf, HH, 0, kt * 32, lane),
               ld_frag_f32(W, 256, wave * 16, kt * 32, lane), a);
#pragma unroll
  for (int kt = 0; kt < 4; ++kt)
    a = wmma32(ld_frag_bf(cbf, HH, 0, kt * 32, lane),
               ld_frag_f32(W + HH, 256, wave * 16, kt * 32, lane), a);
#pragma unroll
  for (int j = 0; j < 8; ++j) e_s[(mb + j) * HH + wave * 16 + nl] = a[j];
}

// One LSTM step for the 16-row tile: G = xa@Wih.T + h@Whh.T + bias (+rank1), then
// gate-aligned activation in registers. Wave w owns hidden cols [16w,16w+16).
template <int XK, bool RANK1, bool STORE>
__device__ __forceinline__ void lstm_step(
    const __bf16* xa, const __bf16* wih, int wih_stride,
    const __bf16* hbf, const __bf16* whh,
    const float* bsum, const float* wlast, const float* ls,
    float* sh_h, float* sh_c, __bf16* sh_hbf, __bf16* sh_cbf,
    float* store_base, int t, int bG0, int lane, int wave)
{
  const int nl = lane & 15;
  const int mb = (lane >> 4) << 3;
  v16bf afr[XK];
#pragma unroll
  for (int kt = 0; kt < XK; ++kt) afr[kt] = ld_frag_bf(xa, HH, 0, kt * 32, lane);
  v16bf hfr[4];
#pragma unroll
  for (int kt = 0; kt < 4; ++kt) hfr[kt] = ld_frag_bf(hbf, HH, 0, kt * 32, lane);

  v8f acc[4];
#pragma unroll
  for (int q = 0; q < 4; ++q) {
    const int n0 = q * HH + wave * 16;
    const float bv = bsum[n0 + nl];
    v8f a;
#pragma unroll
    for (int j = 0; j < 8; ++j) a[j] = bv;
#pragma unroll
    for (int kt = 0; kt < XK; ++kt)
      a = wmma32(afr[kt], ld_frag_bf(wih, wih_stride, n0, kt * 32, lane), a);
#pragma unroll
    for (int kt = 0; kt < 4; ++kt)
      a = wmma32(hfr[kt], ld_frag_bf(whh, HH, n0, kt * 32, lane), a);
    acc[q] = a;
  }

  const int col = wave * 16 + nl;
#pragma unroll
  for (int j = 0; j < 8; ++j) {
    const int mr = mb + j;
    float iv = acc[0][j], fv = acc[1][j], gv = acc[2][j], ov = acc[3][j];
    if constexpr (RANK1) {   // last column of Wih2 (label feature), rank-1 outer product
      const float lw = ls[mr];
      iv += lw * wlast[col];
      fv += lw * wlast[HH + col];
      gv += lw * wlast[2 * HH + col];
      ov += lw * wlast[3 * HH + col];
    }
    const float cn = sigm(fv) * sh_c[mr * HH + col] + sigm(iv) * htanh(gv);
    const float hn = sigm(ov) * htanh(cn);
    sh_c[mr * HH + col] = cn;
    sh_h[mr * HH + col] = hn;
    sh_hbf[mr * HH + col] = (__bf16)hn;
    sh_cbf[mr * HH + col] = (__bf16)cn;
    if constexpr (STORE)
      store_base[(size_t)(bG0 + mr) * (TE * HH) + t * HH + col] = hn;
  }
}

__global__ __launch_bounds__(256, 1) void dstp_rnn_kernel(P p)
{
  __shared__ __align__(16) char smem[SMEM_SZ];
  __bf16* whh_s  = (__bf16*)(smem + OFF_WHH);
  __bf16* wih_s  = (__bf16*)(smem + OFF_WIH);
  float*  wix_s  = (float*)(smem + OFF_WIX);
  float*  bsum_s = (float*)(smem + OFF_BSUM);
  float*  wlast_s= (float*)(smem + OFF_WLAST);
  float*  h_s    = (float*)(smem + OFF_H);
  float*  c_s    = (float*)(smem + OFF_C);
  __bf16* hbf_s  = (__bf16*)(smem + OFF_HBF);
  __bf16* xa_s   = (__bf16*)(smem + OFF_XABF);
  float*  e_s    = (float*)(smem + OFF_E);
  float*  att_s  = (float*)(smem + OFF_ATT);
  float*  ls_s   = (float*)(smem + OFF_LS);
  __bf16* cbf_s  = (__bf16*)(smem + OFF_CBF);
  float*  e2_s   = (float*)(smem + OFF_E2);

  const int tid  = threadIdx.x;
  const int lane = tid & 31;
  const int wave = tid >> 5;
  const int bG0  = blockIdx.x * TB;

  // ---- init state, zero-pad xa fragment buffer
  for (int i = tid; i < TB * HH; i += 256) {
    h_s[i] = 0.f; c_s[i] = 0.f;
    hbf_s[i] = (__bf16)0.f; cbf_s[i] = (__bf16)0.f; xa_s[i] = (__bf16)0.f;
  }

  // ---- phase A staging: Whh1, Wih1 (pad 17->32), biases, WiX tile
  for (int i = tid; i < G4 * HH; i += 256) whh_s[i] = (__bf16)p.Whh1[i];
  for (int i = tid; i < G4 * 32; i += 256) { int n = i >> 5, k = i & 31; wih_s[i] = (__bf16)(k < 17 ? p.Wih1[n * 17 + k] : 0.f); }
  for (int i = tid; i < G4; i += 256) bsum_s[i] = p.bih1[i] + p.bhh1[i];
  for (int idx = tid; idx < TB * 17 * TE; idx += 256) {
    int b = idx / (17 * TE); int r = idx - b * 17 * TE; int f = r / TE; int tt = r - f * TE;
    int col = (f < 14) ? f : f + 1;                 // COLS skips column 14
    const float* xp = p.inp + (size_t)(bG0 + b) * TE * 18 + col;
    const float* wr = p.Wi_w + tt * TE;
    float s = p.Wi_b[tt];
    for (int t2 = 0; t2 < TE; ++t2) s += xp[t2 * 18] * wr[t2];
    wix_s[(b * 17 + f) * TE + tt] = s;
  }
  __syncthreads();

  const float vdb = p.Vd_b[0];

  // ==== phase A: encoder layer 1 (64 steps) ====
  for (int t = 0; t < TE; ++t) {
    attn_e_enc(p.We_w, hbf_s, cbf_s, e_s, e2_s, lane, wave);   // e = [h,c]@We.T (WMMA)
    __syncthreads();
    for (int idx = tid; idx < TB * 17; idx += 256) {           // score over 17 features
      int b = idx / 17, f = idx - b * 17;
      const float* wx = wix_s + (b * 17 + f) * TE;
      float s = vdb;
      for (int tt = 0; tt < TE; ++tt)
        s += htanh(wx[tt] + e_s[b * HH + tt] + e2_s[b * TE + tt]) * p.Vd_w[tt];
      att_s[b * 132 + f] = s;
    }
    __syncthreads();
    if (tid < TB) {                                            // softmax + scaled x -> bf16 A frag
      int b = tid;
      float mx = -1e30f;
      for (int f = 0; f < 17; ++f) mx = fmaxf(mx, att_s[b * 132 + f]);
      float sm = 0.f;
      for (int f = 0; f < 17; ++f) { float eo = __expf(att_s[b * 132 + f] - mx); att_s[b * 132 + f] = eo; sm += eo; }
      float inv = 1.f / sm;
      const float* xr = p.inp + (size_t)(bG0 + b) * TE * 18 + t * 18;
      for (int f = 0; f < 17; ++f) {
        int col = (f < 14) ? f : f + 1;
        xa_s[b * HH + f] = (__bf16)(xr[col] * att_s[b * 132 + f] * inv);
      }
    }
    __syncthreads();
    lstm_step<1, false, true>(xa_s, wih_s, 32, hbf_s, whh_s, bsum_s, nullptr, nullptr,
                              h_s, c_s, hbf_s, cbf_s, p.mid, t, bG0, lane, wave);
    __syncthreads();
  }

  __threadfence(); __syncthreads();

  // ---- bridge: Wi2X[b][j][t] = mid2.T @ Wi2_w.T + Wi2_b (to ws)
  for (int idx = tid; idx < TB * 129 * TE; idx += 256) {
    int b = idx / (129 * TE); int r = idx - b * (129 * TE); int j = r / TE; int tt = r - j * TE;
    const float* wr = p.Wi2_w + tt * TE;
    float s = p.Wi2_b[tt];
    if (j < HH) {
      const float* m = p.mid + (size_t)(bG0 + b) * TE * HH + j;
      for (int t2 = 0; t2 < TE; ++t2) s += m[t2 * HH] * wr[t2];
    } else {
      const float* lp = p.label + (size_t)(bG0 + b) * TE;
      for (int t2 = 0; t2 < TE; ++t2) s += lp[t2] * wr[t2];
    }
    p.wi2x[(size_t)(bG0 + b) * 129 * TE + j * TE + tt] = s;
  }
  __threadfence(); __syncthreads();

  // ---- phase B staging: Whh2, Wih2[:, :128] + last col, biases; reset state
  for (int i = tid; i < G4 * HH; i += 256) whh_s[i] = (__bf16)p.Whh2[i];
  for (int i = tid; i < G4 * HH; i += 256) { int n = i >> 7, k = i & 127; wih_s[i] = (__bf16)p.Wih2[n * 129 + k]; }
  for (int i = tid; i < G4; i += 256) { bsum_s[i] = p.bih2[i] + p.bhh2[i]; wlast_s[i] = p.Wih2[i * 129 + 128]; }
  for (int i = tid; i < TB * HH; i += 256) { h_s[i] = 0.f; c_s[i] = 0.f; hbf_s[i] = (__bf16)0.f; cbf_s[i] = (__bf16)0.f; }
  __syncthreads();

  const float vdb2 = p.Vd2_b[0];

  // ==== phase B: encoder layer 2 (64 steps) ====
  for (int t = 0; t < TE; ++t) {
    attn_e_enc(p.We2_w, hbf_s, cbf_s, e_s, e2_s, lane, wave);  // e = [h,c]@We2.T (WMMA)
    __syncthreads();
    for (int idx = tid; idx < TB * 129; idx += 256) {
      int b = idx / 129, j = idx - b * 129;
      const float* wx = p.wi2x + (size_t)(bG0 + b) * 129 * TE + j * TE;
      float s = vdb2;
      for (int tt = 0; tt < TE; ++tt)
        s += htanh(wx[tt] + e_s[b * HH + tt] + e2_s[b * TE + tt]) * p.Vd2_w[tt];
      att_s[b * 132 + j] = s;
    }
    __syncthreads();
    if (tid < TB) {
      int b = tid;
      float mx = -1e30f;
      for (int j = 0; j < 129; ++j) mx = fmaxf(mx, att_s[b * 132 + j]);
      float sm = 0.f;
      for (int j = 0; j < 129; ++j) { float eo = __expf(att_s[b * 132 + j] - mx); att_s[b * 132 + j] = eo; sm += eo; }
      float inv = 1.f / sm;
      for (int j = 0; j < 129; ++j) att_s[b * 132 + j] *= inv;
    }
    __syncthreads();
    for (int idx = tid; idx < TB * HH; idx += 256) {           // attention-scaled x (first 128 feats)
      int b = idx >> 7, j = idx & 127;
      xa_s[idx] = (__bf16)(p.mid[(size_t)(bG0 + b) * TE * HH + t * HH + j] * att_s[b * 132 + j]);
    }
    if (tid < TB) ls_s[tid] = p.label[(size_t)(bG0 + tid) * TE + t] * att_s[tid * 132 + 128];
    __syncthreads();
    lstm_step<4, true, true>(xa_s, wih_s, HH, hbf_s, whh_s, bsum_s, wlast_s, ls_s,
                             h_s, c_s, hbf_s, cbf_s, p.fin, t, bG0, lane, wave);
    __syncthreads();
  }

  __threadfence(); __syncthreads();

  // ---- bridge: WxF = final @ Wx_w.T + Wx_b, via WMMA (f32->bf16 fragment loads)
  {
    const int nl = lane & 15; const int mb = (lane >> 4) << 3;
    for (int tile = wave; tile < TE * 8; tile += 8) {
      const int t = tile >> 3; const int n0 = (tile & 7) * 16;
      const float bv = p.Wx_b[n0 + nl];
      v8f a;
#pragma unroll
      for (int j = 0; j < 8; ++j) a[j] = bv;
#pragma unroll
      for (int kt = 0; kt < 4; ++kt) {
        v16bf af = ld_frag_f32(p.fin + (size_t)bG0 * TE * HH + t * HH, TE * HH, 0, kt * 32, lane);
        v16bf bf = ld_frag_f32(p.Wx_w, HH, n0, kt * 32, lane);
        a = wmma32(af, bf, a);
      }
#pragma unroll
      for (int j = 0; j < 8; ++j)
        p.wxf[(size_t)(bG0 + mb + j) * TE * HH + t * HH + n0 + nl] = a[j];
    }
  }
  __threadfence(); __syncthreads();

  // ---- phase C staging: Whhd, Wihd, biases; reset state
  for (int i = tid; i < G4 * HH; i += 256) { whh_s[i] = (__bf16)p.Whhd[i]; wih_s[i] = (__bf16)p.Wihd[i]; }
  for (int i = tid; i < G4; i += 256) bsum_s[i] = p.bihd[i] + p.bhhd[i];
  for (int i = tid; i < TB * HH; i += 256) { h_s[i] = 0.f; c_s[i] = 0.f; hbf_s[i] = (__bf16)0.f; cbf_s[i] = (__bf16)0.f; }
  __syncthreads();

  const float vb  = p.V_b[0];
  const float rgb = p.reg_b[0];

  // ==== phase C: decoder (24 steps) ====
  for (int s = 0; s < TDTOT; ++s) {
    attn_e_dec(p.Wh_w, hbf_s, cbf_s, e_s, lane, wave);         // e = [h,c]@Wh.T (WMMA, N=128)
    __syncthreads();
    for (int idx = tid; idx < TB * TE; idx += 256) {           // temporal scores
      int b = idx >> 6, tt = idx & 63;
      const float* w = p.wxf + (size_t)(bG0 + b) * TE * HH + tt * HH;
      float acc = vb;
      for (int n = 0; n < HH; ++n) acc += htanh(w[n] + e_s[b * HH + n]) * p.V_w[n];
      att_s[b * 132 + tt] = acc;
    }
    __syncthreads();
    if (tid < TB) {
      int b = tid;
      float mx = -1e30f;
      for (int tt = 0; tt < TE; ++tt) mx = fmaxf(mx, att_s[b * 132 + tt]);
      float sm = 0.f;
      for (int tt = 0; tt < TE; ++tt) { float eo = __expf(att_s[b * 132 + tt] - mx); att_s[b * 132 + tt] = eo; sm += eo; }
      float inv = 1.f / sm;
      for (int tt = 0; tt < TE; ++tt) att_s[b * 132 + tt] *= inv;
    }
    __syncthreads();
    for (int idx = tid; idx < TB * HH; idx += 256) {           // din = a . final -> bf16 A frag
      int b = idx >> 7, n = idx & 127;
      const float* f = p.fin + (size_t)(bG0 + b) * TE * HH + n;
      float acc = 0.f;
      for (int tt = 0; tt < TE; ++tt) acc += att_s[b * 132 + tt] * f[tt * HH];
      xa_s[idx] = (__bf16)acc;
    }
    __syncthreads();
    lstm_step<4, false, false>(xa_s, wih_s, HH, hbf_s, whh_s, bsum_s, nullptr, nullptr,
                               h_s, c_s, hbf_s, cbf_s, nullptr, 0, bG0, lane, wave);
    __syncthreads();
    if (s >= 6 && tid < TB) {                                  // regression head
      float o = rgb;
      for (int n = 0; n < HH; ++n) o += h_s[tid * HH + n] * p.reg_w[n];
      p.out[(size_t)(bG0 + tid) * 18 + (s - 6)] = o;
    }
    __syncthreads();
  }
}

extern "C" void kernel_launch(void* const* d_in, const int* in_sizes, int n_in,
                              void* d_out, int out_size, void* d_ws, size_t ws_size,
                              hipStream_t stream) {
  P p;
  p.inp   = (const float*)d_in[0];   p.label = (const float*)d_in[1];
  p.Wih1  = (const float*)d_in[2];   p.Whh1  = (const float*)d_in[3];
  p.bih1  = (const float*)d_in[4];   p.bhh1  = (const float*)d_in[5];
  p.Wih2  = (const float*)d_in[6];   p.Whh2  = (const float*)d_in[7];
  p.bih2  = (const float*)d_in[8];   p.bhh2  = (const float*)d_in[9];
  p.Wihd  = (const float*)d_in[10];  p.Whhd  = (const float*)d_in[11];
  p.bihd  = (const float*)d_in[12];  p.bhhd  = (const float*)d_in[13];
  p.Wi_w  = (const float*)d_in[14];  p.Wi_b  = (const float*)d_in[15];
  p.We_w  = (const float*)d_in[16];  p.Vd_w  = (const float*)d_in[17];
  p.Vd_b  = (const float*)d_in[18];  p.Wi2_w = (const float*)d_in[19];
  p.Wi2_b = (const float*)d_in[20];  p.We2_w = (const float*)d_in[21];
  p.Vd2_w = (const float*)d_in[22];  p.Vd2_b = (const float*)d_in[23];
  p.Wx_w  = (const float*)d_in[24];  p.Wx_b  = (const float*)d_in[25];
  p.Wh_w  = (const float*)d_in[26];  p.V_w   = (const float*)d_in[27];
  p.V_b   = (const float*)d_in[28];  p.reg_w = (const float*)d_in[29];
  p.reg_b = (const float*)d_in[30];

  float* ws = (float*)d_ws;
  p.mid  = ws;                                      // 512*64*128
  p.wi2x = p.mid  + (size_t)512 * 64 * 128;         // 512*129*64
  p.fin  = p.wi2x + (size_t)512 * 129 * 64;         // 512*64*128
  p.wxf  = p.fin  + (size_t)512 * 64 * 128;         // 512*64*128
  p.out  = (float*)d_out;

  dstp_rnn_kernel<<<dim3(512 / TB), dim3(256), 0, stream>>>(p);
}
